// baselineGIN_35553739276822
// MI455X (gfx1250) — matrix operand order repaired and checked by reading, compiled, not run on hardware
//
#include <hip/hip_runtime.h>

#define Hc   128
#define ROWS 64
#define TPB  256

typedef __attribute__((ext_vector_type(16))) _Float16 v16h;
typedef __attribute__((ext_vector_type(8)))  _Float16 v8h;
typedef __attribute__((ext_vector_type(8)))  float    v8f;

// A-fragment (16x32 f16, M x K) per ISA 7.12.2:
// lanes 0-15: row M=lane, K = ks*32 + {0..7, 16..23}
// lanes 16-31: row M=lane-16, K = ks*32 + {8..15, 24..31}
__device__ __forceinline__ v16h frag_a(const _Float16* tile, int row0, int ks, int lane) {
  int r  = lane & 15;
  int kb = ks * 32 + ((lane >> 4) << 3);
  const _Float16* p = tile + (row0 + r) * Hc + kb;
  v8h lo = *(const v8h*)p;          // K kb..kb+7
  v8h hi = *(const v8h*)(p + 16);   // K kb+16..kb+23
  return __builtin_shufflevector(lo, hi, 0,1,2,3,4,5,6,7,8,9,10,11,12,13,14,15);
}

// B-fragment (32x16 f16, K x N), weights stored column-major in LDS (wt[n*128+k]):
// lanes 0-15: col N=lane, K = ks*32 + {0..7, 16..23} ; lanes 16-31: col N=lane-16, K = ks*32 + {8..15, 24..31}
__device__ __forceinline__ v16h frag_b(const _Float16* wt, int col0, int ks, int lane) {
  int c  = lane & 15;
  int kb = ks * 32 + ((lane >> 4) << 3);
  const _Float16* p = wt + (col0 + c) * Hc + kb;
  v8h lo = *(const v8h*)p;
  v8h hi = *(const v8h*)(p + 16);
  return __builtin_shufflevector(lo, hi, 0,1,2,3,4,5,6,7,8,9,10,11,12,13,14,15);
}

// -------- embedding gather: h[n,c] = table[x[n], c] --------
__global__ __launch_bounds__(TPB) void gin_embed(const int* __restrict__ x,
                                                 const float* __restrict__ table,
                                                 float* __restrict__ h, int n) {
  size_t gid = (size_t)blockIdx.x * TPB + threadIdx.x;
  if (gid >= (size_t)n * Hc) return;
  int node = (int)(gid >> 7), c = (int)(gid & 127);
  h[gid] = table[(size_t)x[node] * Hc + c];
}

// -------- layer stage 1: y = relu(BN(h) @ W + bias); agg = 0 --------
__global__ __launch_bounds__(TPB) void gin_lin(
    const float* __restrict__ hin, const float* __restrict__ W,
    const float* __restrict__ bias,
    const float* __restrict__ g, const float* __restrict__ b,
    const float* __restrict__ m, const float* __restrict__ v,
    float* __restrict__ yout, float* __restrict__ aggz, int n) {
  __shared__ __align__(16) _Float16 Wt[Hc * Hc];
  __shared__ __align__(16) _Float16 A[ROWS * Hc];
  __shared__ float sscale[Hc], sshift[Hc], sbias[Hc];

  int tid = threadIdx.x, lane = tid & 31, wave = tid >> 5;
  for (int idx = tid; idx < Hc * Hc; idx += TPB) {
    int k = idx >> 7, c = idx & 127;
    Wt[c * Hc + k] = (_Float16)W[idx];
  }
  if (tid < Hc) {
    float sc = g[tid] * rsqrtf(v[tid] + 1e-5f);
    sscale[tid] = sc;
    sshift[tid] = b[tid] - m[tid] * sc;
    sbias[tid]  = bias[tid];
  }
  __syncthreads();

  int row0 = blockIdx.x * ROWS;
  for (int idx = tid; idx < ROWS * Hc; idx += TPB) {
    int r = idx >> 7, c = idx & 127;
    int gr = row0 + r;
    float xv = (gr < n) ? hin[(size_t)gr * Hc + c] : 0.0f;
    A[idx] = (_Float16)(xv * sscale[c] + sshift[c]);
  }
  __syncthreads();

  v8f acc[4] = {};
  for (int ks = 0; ks < 4; ++ks) {
    v16h bf = frag_b(Wt, wave * 16, ks, lane);
    for (int mt = 0; mt < 4; ++mt) {
      v16h af = frag_a(A, mt * 16, ks, lane);
      acc[mt] = __builtin_amdgcn_wmma_f32_16x16x32_f16(
          false, af, false, bf, (short)0, acc[mt], false, false);
    }
  }

  int c = wave * 16 + (lane & 15);
  int rbase = (lane >> 4) << 3;
  for (int mt = 0; mt < 4; ++mt)
    for (int vv = 0; vv < 8; ++vv) {
      int gr = row0 + mt * 16 + rbase + vv;
      if (gr < n) {
        float t = acc[mt][vv] + sbias[c];
        t = t > 0.0f ? t : 0.0f;
        yout[(size_t)gr * Hc + c] = t;
        aggz[(size_t)gr * Hc + c] = 0.0f;
      }
    }
}

// -------- edge scatter: agg[dst] += y[src]; 32 lanes/edge, float4 per lane --------
__global__ __launch_bounds__(TPB) void gin_scatter(const int* __restrict__ src,
                                                   const int* __restrict__ dst,
                                                   const float* __restrict__ y,
                                                   float* __restrict__ agg, int e) {
  long long gid = (long long)blockIdx.x * TPB + threadIdx.x;
  int ed = (int)(gid >> 5);
  if (ed >= e) return;
  int q = (int)(gid & 31) * 4;
  int s = src[ed], d = dst[ed];
  float4 val = *(const float4*)(y + (size_t)s * Hc + q);
  float* ap = agg + (size_t)d * Hc + q;
  atomicAdd(ap + 0, val.x);
  atomicAdd(ap + 1, val.y);
  atomicAdd(ap + 2, val.z);
  atomicAdd(ap + 3, val.w);
}

// -------- layer stage 2 (fused MLP): h = relu(BN((1+eps)y+agg) @W1 +b1)) @ W2 + b2 --------
__global__ __launch_bounds__(TPB) void gin_mlp(
    const float* __restrict__ y, const float* __restrict__ agg,
    const float* __restrict__ epsp,
    const float* __restrict__ W1, const float* __restrict__ b1,
    const float* __restrict__ mg, const float* __restrict__ mb,
    const float* __restrict__ mm, const float* __restrict__ mv,
    const float* __restrict__ W2, const float* __restrict__ b2,
    float* __restrict__ hout, int n) {
  __shared__ __align__(16) _Float16 W1t[Hc * Hc];
  __shared__ __align__(16) _Float16 W2t[Hc * Hc];
  __shared__ __align__(16) _Float16 A[ROWS * Hc];
  __shared__ __align__(16) _Float16 Z[ROWS * Hc];
  __shared__ float sc1[Hc], off1[Hc], sb2[Hc];

  int tid = threadIdx.x, lane = tid & 31, wave = tid >> 5;
  for (int idx = tid; idx < Hc * Hc; idx += TPB) {
    int k = idx >> 7, c = idx & 127;
    W1t[c * Hc + k] = (_Float16)W1[idx];
    W2t[c * Hc + k] = (_Float16)W2[idx];
  }
  if (tid < Hc) {
    float sc = mg[tid] * rsqrtf(mv[tid] + 1e-5f);
    sc1[tid]  = sc;
    off1[tid] = b1[tid] * sc + (mb[tid] - mm[tid] * sc);
    sb2[tid]  = b2[tid];
  }
  float e1 = 1.0f + epsp[0];

  int row0 = blockIdx.x * ROWS;
  for (int idx = tid; idx < ROWS * Hc; idx += TPB) {
    int r = idx >> 7, c = idx & 127;
    int gr = row0 + r;
    float zv = 0.0f;
    if (gr < n) {
      size_t o = (size_t)gr * Hc + c;
      zv = e1 * y[o] + agg[o];
    }
    A[idx] = (_Float16)zv;
  }
  __syncthreads();

  int c = wave * 16 + (lane & 15);
  int rbase = (lane >> 4) << 3;

  // GEMM1: A @ W1 -> Z (with BN fold + relu)
  {
    v8f acc[4] = {};
    for (int ks = 0; ks < 4; ++ks) {
      v16h bf = frag_b(W1t, wave * 16, ks, lane);
      for (int mt = 0; mt < 4; ++mt) {
        v16h af = frag_a(A, mt * 16, ks, lane);
        acc[mt] = __builtin_amdgcn_wmma_f32_16x16x32_f16(
            false, af, false, bf, (short)0, acc[mt], false, false);
      }
    }
    for (int mt = 0; mt < 4; ++mt)
      for (int vv = 0; vv < 8; ++vv) {
        int rl = mt * 16 + rbase + vv;
        float t = acc[mt][vv] * sc1[c] + off1[c];
        t = t > 0.0f ? t : 0.0f;
        Z[rl * Hc + c] = (_Float16)t;
      }
  }
  __syncthreads();

  // GEMM2: Z @ W2 + b2 -> hout
  {
    v8f acc[4] = {};
    for (int ks = 0; ks < 4; ++ks) {
      v16h bf = frag_b(W2t, wave * 16, ks, lane);
      for (int mt = 0; mt < 4; ++mt) {
        v16h af = frag_a(Z, mt * 16, ks, lane);
        acc[mt] = __builtin_amdgcn_wmma_f32_16x16x32_f16(
            false, af, false, bf, (short)0, acc[mt], false, false);
      }
    }
    for (int mt = 0; mt < 4; ++mt)
      for (int vv = 0; vv < 8; ++vv) {
        int gr = row0 + mt * 16 + rbase + vv;
        if (gr < n) hout[(size_t)gr * Hc + c] = acc[mt][vv] + sb2[c];
      }
  }
}

// -------- zero the output --------
__global__ __launch_bounds__(TPB) void gin_zero(float* __restrict__ out, int n) {
  int gid = blockIdx.x * TPB + threadIdx.x;
  if (gid < n) out[gid] = 0.0f;
}

// -------- readout: out[batch[n]] += h[n] . ro_w + ro_b  (wave per node) --------
__global__ __launch_bounds__(TPB) void gin_readout(const float* __restrict__ h,
                                                   const int* __restrict__ batch,
                                                   const float* __restrict__ rw,
                                                   const float* __restrict__ rb,
                                                   float* __restrict__ out, int n) {
  long long gid = (long long)blockIdx.x * TPB + threadIdx.x;
  int node = (int)(gid >> 5);
  if (node >= n) return;
  int lane = threadIdx.x & 31;
  float4 hv = *(const float4*)(h + (size_t)node * Hc + lane * 4);
  float4 wv = *(const float4*)(rw + lane * 4);
  float p = hv.x * wv.x + hv.y * wv.y + hv.z * wv.z + hv.w * wv.w;
  for (int o = 16; o > 0; o >>= 1) p += __shfl_xor(p, o, 32);
  if (lane == 0) atomicAdd(&out[batch[node]], p + rb[0]);
}

extern "C" void kernel_launch(void* const* d_in, const int* in_sizes, int n_in,
                              void* d_out, int out_size, void* d_ws, size_t ws_size,
                              hipStream_t stream) {
  (void)n_in; (void)ws_size;
  const int*   x     = (const int*)d_in[0];
  const int*   ei    = (const int*)d_in[1];
  const int*   batch = (const int*)d_in[2];
  const float* table = (const float*)d_in[3];
  const float* bng   = (const float*)d_in[4];
  const float* bnb   = (const float*)d_in[5];
  const float* bnm   = (const float*)d_in[6];
  const float* bnv   = (const float*)d_in[7];
  const float* linw  = (const float*)d_in[8];
  const float* linb  = (const float*)d_in[9];
  const float* eps   = (const float*)d_in[10];
  const float* w1    = (const float*)d_in[11];
  const float* mb1   = (const float*)d_in[12];
  const float* mbg   = (const float*)d_in[13];
  const float* mbb   = (const float*)d_in[14];
  const float* mbm   = (const float*)d_in[15];
  const float* mbv   = (const float*)d_in[16];
  const float* w2    = (const float*)d_in[17];
  const float* mb2   = (const float*)d_in[18];
  const float* rw    = (const float*)d_in[19];
  const float* rb    = (const float*)d_in[20];

  int Nn = in_sizes[0];
  int Ee = in_sizes[1] / 2;
  size_t nh = (size_t)Nn * Hc;

  float* h  = (float*)d_ws;   // current node features
  float* yb = h + nh;         // post-lin activations
  float* ag = yb + nh;        // edge aggregation

  dim3 blk(TPB);
  int gEmbed = (int)((nh + TPB - 1) / TPB);
  gin_embed<<<gEmbed, blk, 0, stream>>>(x, table, h, Nn);

  int gRows = (Nn + ROWS - 1) / ROWS;
  long long sthr = (long long)Ee * 32;
  int gScat = (int)((sthr + TPB - 1) / TPB);

  for (int i = 0; i < 3; ++i) {
    gin_lin<<<gRows, blk, 0, stream>>>(h, linw + (size_t)i * Hc * Hc, linb + i * Hc,
                                       bng + i * Hc, bnb + i * Hc, bnm + i * Hc, bnv + i * Hc,
                                       yb, ag, Nn);
    gin_scatter<<<gScat, blk, 0, stream>>>(ei, ei + Ee, yb, ag, Ee);
    gin_mlp<<<gRows, blk, 0, stream>>>(yb, ag, eps + i,
                                       w1 + (size_t)i * Hc * Hc, mb1 + i * Hc,
                                       mbg + i * Hc, mbb + i * Hc, mbm + i * Hc, mbv + i * Hc,
                                       w2 + (size_t)i * Hc * Hc, mb2 + i * Hc, h, Nn);
  }

  gin_zero<<<(out_size + TPB - 1) / TPB, blk, 0, stream>>>((float*)d_out, out_size);
  long long rthr = (long long)Nn * 32;
  gin_readout<<<(int)((rthr + TPB - 1) / TPB), blk, 0, stream>>>(h, batch, rw, rb,
                                                                 (float*)d_out, Nn);
}
